// GenConv_24721831755817
// MI455X (gfx1250) — matrix operand order repaired and checked by reading, not compiled, not run on hardware
//
#include <hip/hip_runtime.h>
#include <math.h>

#define N_NODES 50000
#define N_EDGES 800000
#define IN_C    16
#define OUT_C   16
#define POS_C   16
#define K_OFF   8
#define TEMP    8.0f

typedef __attribute__((ext_vector_type(16))) _Float16 v16h;
typedef __attribute__((ext_vector_type(8)))  _Float16 h8;
typedef __attribute__((ext_vector_type(8)))  float    v8f;

// out[n,o] = bias[o]  (covers nodes with no incoming edges; edge kernel atomically adds)
__global__ void genconv_bias_init(float* __restrict__ out,
                                  const float* __restrict__ bias,
                                  int total) {
    int idx = blockIdx.x * blockDim.x + threadIdx.x;
    if (idx < total) out[idx] = bias[idx & (OUT_C - 1)];
}

__global__ __launch_bounds__(256)
void genconv_edge_kernel(const float* __restrict__ x,
                         const float* __restrict__ pos,
                         const int*   __restrict__ edge_index,
                         const float* __restrict__ edge_attr,
                         const float* __restrict__ offset,
                         const float* __restrict__ weight,
                         float* __restrict__ out)
{
    // Per-edge A rows: 256 edges/block * 128 halves (K-major: K = 16*k + i)
    __shared__ __align__(16) _Float16 ldsA[256 * 128];
    __shared__ __align__(16) float ldsOff[K_OFF * POS_C];
    __shared__ float ldsO2[K_OFF];

    const int tid  = threadIdx.x;
    const int lane = tid & 31;
    const int wave = tid >> 5;
    const int o    = lane & 15;          // output channel owned by this lane (B/C/D column)
    const int hi   = (lane >> 4) & 1;    // lane in upper half-wave?
    const int blockBase = blockIdx.x * 256;
    const int e = blockBase + tid;       // edge owned by this lane in the scalar phase

    // ---- stage offsets + ||offset_k||^2 in LDS ----
    if (tid < K_OFF * POS_C) ldsOff[tid] = offset[tid];
    __syncthreads();
    if (tid < K_OFF) {
        float s = 0.f;
        #pragma unroll
        for (int i = 0; i < POS_C; ++i) { float v = ldsOff[tid * POS_C + i]; s += v * v; }
        ldsO2[tid] = s;
    }
    __syncthreads();

    // ---- per-edge scalar phase: d, sim, softmax alpha, gather x[col] ----
    float alpha[K_OFF];
    float xc[IN_C];
    if (e < N_EDGES) {
        const int r = edge_index[e];
        const int c = edge_index[N_EDGES + e];
        float d[POS_C];
        const float4* pos4 = (const float4*)pos;
        #pragma unroll
        for (int q = 0; q < 4; ++q) {
            float4 pr = pos4[r * 4 + q];
            float4 pc = pos4[c * 4 + q];
            d[4*q+0] = pc.x - pr.x; d[4*q+1] = pc.y - pr.y;
            d[4*q+2] = pc.z - pr.z; d[4*q+3] = pc.w - pr.w;
        }
        float d2 = 0.f;
        #pragma unroll
        for (int i = 0; i < POS_C; ++i) d2 += d[i] * d[i];

        float mx = -1e30f;
        #pragma unroll
        for (int k = 0; k < K_OFF; ++k) {
            float dot = 0.f;
            #pragma unroll
            for (int i = 0; i < POS_C; ++i) dot += d[i] * ldsOff[k * POS_C + i];
            float sq = d2 + ldsO2[k] - 2.0f * dot;
            sq = fmaxf(sq, 0.0f);
            float sim = -sqrtf(sq) * TEMP;
            alpha[k] = sim;
            mx = fmaxf(mx, sim);
        }
        float sum = 0.f;
        #pragma unroll
        for (int k = 0; k < K_OFF; ++k) { alpha[k] = __expf(alpha[k] - mx); sum += alpha[k]; }
        float inv = 1.0f / sum;
        #pragma unroll
        for (int k = 0; k < K_OFF; ++k) alpha[k] *= inv;

        const float4* x4 = (const float4*)x;
        #pragma unroll
        for (int q = 0; q < 4; ++q) {
            float4 v = x4[c * 4 + q];
            xc[4*q+0] = v.x; xc[4*q+1] = v.y; xc[4*q+2] = v.z; xc[4*q+3] = v.w;
        }
    } else {
        #pragma unroll
        for (int k = 0; k < K_OFF; ++k) alpha[k] = 0.f;
        #pragma unroll
        for (int i = 0; i < IN_C; ++i)  xc[i] = 0.f;
    }

    // ---- stage A row: A[e, 16*k + i] = f16(alpha_k * x_i) ----
    _Float16* myA = &ldsA[tid * 128];
    #pragma unroll
    for (int k = 0; k < K_OFF; ++k) {
        h8 v0, v1;
        #pragma unroll
        for (int i = 0; i < 8; ++i) {
            v0[i] = (_Float16)(alpha[k] * xc[i]);
            v1[i] = (_Float16)(alpha[k] * xc[8 + i]);
        }
        *(h8*)(myA + k * 16)     = v0;
        *(h8*)(myA + k * 16 + 8) = v1;
    }
    __syncthreads();

    // ---- B fragments (per-lane constant): chunk c -> k_off = 2c + hi, element h -> i = h ----
    v16h bfrag[4];
    #pragma unroll
    for (int c = 0; c < 4; ++c) {
        const int koff = 2 * c + hi;
        const float* wrow = weight + koff * (OUT_C * IN_C) + o * IN_C;
        #pragma unroll
        for (int h = 0; h < 16; ++h) bfrag[c][h] = (_Float16)wrow[h];
    }

    // ---- two 16-edge WMMA tiles per wave: D[16x16] = A[16x128] * B[128x16] ----
    #pragma unroll
    for (int t = 0; t < 2; ++t) {
        v8f acc = {};
        const int slot = wave * 32 + t * 16 + (lane & 15);   // A row M = lane&15
        const _Float16* arow = &ldsA[slot * 128];
        const int kb0 = hi ? 16 : 0;
        #pragma unroll
        for (int c = 0; c < 4; ++c) {
            const int kb = 32 * c + kb0;
            h8 lo = *(const h8*)(arow + kb);        // K = kb .. kb+7
            h8 hh = *(const h8*)(arow + kb + 8);    // K = kb+8 .. kb+15
            v16h a;
            #pragma unroll
            for (int h = 0; h < 8; ++h) { a[h] = lo[h]; a[h + 8] = hh[h]; }
            acc = __builtin_amdgcn_wmma_f32_16x16x32_f16(
                      false, a, false, bfrag[c], (short)0, acc, false, false);
        }
        // epilogue: VGPR r holds edge m = r + 8*hi, channel o = lane&15
        const int ebase = blockBase + wave * 32 + t * 16;
        #pragma unroll
        for (int r = 0; r < 8; ++r) {
            const int e2 = ebase + r + 8 * hi;
            if (e2 < N_EDGES) {
                const float attr = edge_attr[e2];
                const int   row  = edge_index[e2];
                atomicAdd(&out[row * OUT_C + o], acc[r] * attr);
            }
        }
    }
}

extern "C" void kernel_launch(void* const* d_in, const int* in_sizes, int n_in,
                              void* d_out, int out_size, void* d_ws, size_t ws_size,
                              hipStream_t stream) {
    const float* x        = (const float*)d_in[0];
    const float* pos      = (const float*)d_in[1];
    const int*   eidx     = (const int*)  d_in[2];
    const float* eattr    = (const float*)d_in[3];
    const float* offset   = (const float*)d_in[4];
    const float* weight   = (const float*)d_in[5];
    const float* bias     = (const float*)d_in[6];
    float* out = (float*)d_out;

    const int total = N_NODES * OUT_C;
    genconv_bias_init<<<(total + 255) / 256, 256, 0, stream>>>(out, bias, total);
    genconv_edge_kernel<<<(N_EDGES + 255) / 256, 256, 0, stream>>>(
        x, pos, eidx, eattr, offset, weight, out);
}
